// GRU_66133906424543
// MI455X (gfx1250) — matrix operand order repaired and checked by reading, not compiled, not run on hardware
//
#include <hip/hip_runtime.h>
#include <hip/hip_bf16.h>

// ---------------------------------------------------------------------------
// GRU (S=2048, B=64, I=H=256) for gfx1250.
// Phase 1: convert weights f32 -> bf16 into workspace, zero h state + barrier.
// Phase 2: persistent 16-WG kernel. Each WG owns 4 output tiles (16x16) with a
//   common batch-row group. Waves 0-3: recurrent GEMMs (h_{t-1} W_h^T) + gate
//   math for step t. Waves 4-7: input projections (x_{t+1} W_i^T) into LDS
//   double buffer (overlapped with the recurrence). Device-wide step barrier
//   via an agent-scope atomic counter in L2.
// This revision: each wave preloads its 24 step-invariant weight B-tiles
//   (3 gates x 8 K-tiles, 192 VGPRs) into registers before the loop, removing
//   all weight traffic from the serial recurrence's critical path.
// ---------------------------------------------------------------------------

typedef __bf16 bf16_t;
typedef __attribute__((ext_vector_type(16))) __bf16 v16bf;
typedef __attribute__((ext_vector_type(8)))  __bf16 v8bf;
typedef __attribute__((ext_vector_type(8)))  float  v8f;

#define S_LEN 2048
#define BATCH 64
#define HID   256
#define NWG   16

__device__ __forceinline__ v8f splat8(float v) {
    v8f r;
#pragma unroll
    for (int i = 0; i < 8; ++i) r[i] = v;
    return r;
}

__device__ __forceinline__ float fast_sigmoid(float x) {
    return 1.0f / (1.0f + __expf(-x));
}
__device__ __forceinline__ float fast_tanh(float x) {
    // 1 - 2/(e^{2x}+1): stable at +/- inf
    float e = __expf(2.0f * x);
    return 1.0f - 2.0f / (e + 1.0f);
}

// ---------------------------------------------------------------------------
// Prep: weights -> bf16, zero h buffers (both parity copies), zero counter.
// idx ranges: [0, 6*65536) weights, then 32768 bf16 h (2 buffers), then
// 16384 f32 h. Grid = 442368/256 = 1728 blocks.
// ---------------------------------------------------------------------------
__global__ void gru_prep(const float* __restrict__ Whr, const float* __restrict__ Whz,
                         const float* __restrict__ Whn, const float* __restrict__ Wir,
                         const float* __restrict__ Wiz, const float* __restrict__ Win,
                         bf16_t* __restrict__ wbf, bf16_t* __restrict__ hbf,
                         float* __restrict__ hf, unsigned* __restrict__ ctr)
{
    unsigned idx = blockIdx.x * 256u + threadIdx.x;
    if (idx == 0) *ctr = 0u;
    if (idx < 6u * 65536u) {
        unsigned m = idx >> 16, o = idx & 65535u;
        const float* src = (m == 0) ? Whr : (m == 1) ? Whz : (m == 2) ? Whn
                         : (m == 3) ? Wir : (m == 4) ? Wiz : Win;
        wbf[idx] = (bf16_t)src[o];
    } else if (idx < 6u * 65536u + 32768u) {
        hbf[idx - 6u * 65536u] = (bf16_t)0.0f;
    } else if (idx < 6u * 65536u + 32768u + 16384u) {
        hf[idx - (6u * 65536u + 32768u)] = 0.0f;
    }
}

// ---------------------------------------------------------------------------
// Persistent GRU kernel.
// ---------------------------------------------------------------------------
__global__ void __launch_bounds__(256) gru_main(
    const float*  __restrict__ x,     // (S,B,I)
    const bf16_t* __restrict__ wbf,   // [6][H][K] bf16: Whr,Whz,Whn,Wir,Wiz,Win
    bf16_t*       __restrict__ hbf,   // [2][B][H] bf16 (ping-pong)
    float*        __restrict__ hf,    // [B][H] f32
    unsigned*     __restrict__ ctr,   // step barrier
    const float* __restrict__ b_ir, const float* __restrict__ b_iz,
    const float* __restrict__ b_in, const float* __restrict__ b_hr,
    const float* __restrict__ b_hz, const float* __restrict__ b_hn,
    float* __restrict__ out)          // S*B*H output then B*H h_n
{
    __shared__ v8f buf[2][4][3][32];  // [parity][tile j][gate][lane] = 24 KB

    const int tid   = threadIdx.x;
    const int wave  = tid >> 5;
    const int lane  = tid & 31;
    const int g     = blockIdx.x;     // 0..15
    const int j     = wave & 3;       // tile within WG
    const bool isInput = wave >= 4;

    const int blockId = g * 4 + j;            // 0..63
    const int m0 = (blockId >> 4) * 16;       // batch-row tile base (shared per WG)
    const int n0 = (blockId & 15) * 16;       // hidden-col tile base
    const int lr = lane & 15;
    const int hi = lane >> 4;
    const int Mrow = m0 + lr;                 // A-matrix row for this lane
    const int Ncol = n0 + lr;                 // B/D column for this lane

    // Per-wave gate weight rows + biases (constant across steps).
    const bf16_t *w0, *w1, *w2;
    float bias0, bias1, bias2;
    if (isInput) {
        w0 = wbf + (size_t)3 * 65536 + (size_t)Ncol * HID;  // W_ir
        w1 = wbf + (size_t)4 * 65536 + (size_t)Ncol * HID;  // W_iz
        w2 = wbf + (size_t)5 * 65536 + (size_t)Ncol * HID;  // W_in
        bias0 = b_ir[Ncol]; bias1 = b_iz[Ncol]; bias2 = b_in[Ncol];
    } else {
        w0 = wbf + (size_t)0 * 65536 + (size_t)Ncol * HID;  // W_hr
        w1 = wbf + (size_t)1 * 65536 + (size_t)Ncol * HID;  // W_hz
        w2 = wbf + (size_t)2 * 65536 + (size_t)Ncol * HID;  // W_hn
        bias0 = b_hr[Ncol]; bias1 = b_hz[Ncol]; bias2 = b_hn[Ncol];
    }

    // ---- Preload this wave's 24 weight B-tiles into registers (192 VGPRs).
    // Same registers serve both roles (pointers differ), so peak pressure
    // stays ~250 VGPRs.
    v16bf wt[24];
#pragma unroll
    for (int kt = 0; kt < 8; ++kt) {
        const int kb = kt * 32 + hi * 8;
        wt[0 * 8 + kt] = __builtin_shufflevector(*(const v8bf*)(w0 + kb), *(const v8bf*)(w0 + kb + 16), 0,1,2,3,4,5,6,7,8,9,10,11,12,13,14,15);
        wt[1 * 8 + kt] = __builtin_shufflevector(*(const v8bf*)(w1 + kb), *(const v8bf*)(w1 + kb + 16), 0,1,2,3,4,5,6,7,8,9,10,11,12,13,14,15);
        wt[2 * 8 + kt] = __builtin_shufflevector(*(const v8bf*)(w2 + kb), *(const v8bf*)(w2 + kb + 16), 0,1,2,3,4,5,6,7,8,9,10,11,12,13,14,15);
    }

    // Input projection for time tp -> buf[tp&1][j][*][lane]
    auto inputProj = [&](int tp) {
        const float* xp = x + ((size_t)tp * BATCH + (size_t)Mrow) * HID;
        // prefetch next step's row
        if (tp + 1 < S_LEN)
            __builtin_prefetch(xp + (size_t)BATCH * HID, 0, 1);
        v8f a0 = splat8(bias0), a1 = splat8(bias1), a2 = splat8(bias2);
#pragma unroll
        for (int kt = 0; kt < 8; ++kt) {
            const int ka = kt * 32 + hi * 8;
            v8f f0 = *(const v8f*)(xp + ka);
            v8f f1 = *(const v8f*)(xp + ka + 16);
            v16bf a;
#pragma unroll
            for (int i = 0; i < 8; ++i) { a[i] = (bf16_t)f0[i]; a[i + 8] = (bf16_t)f1[i]; }
            a0 = __builtin_amdgcn_wmma_f32_16x16x32_bf16(false, a, false, wt[0 * 8 + kt], (short)0, a0, false, false);
            a1 = __builtin_amdgcn_wmma_f32_16x16x32_bf16(false, a, false, wt[1 * 8 + kt], (short)0, a1, false, false);
            a2 = __builtin_amdgcn_wmma_f32_16x16x32_bf16(false, a, false, wt[2 * 8 + kt], (short)0, a2, false, false);
        }
        const int pb = tp & 1;
        buf[pb][j][0][lane] = a0;
        buf[pb][j][1][lane] = a1;
        buf[pb][j][2][lane] = a2;
    };

    // Prologue: projections for t=0 into buf[0].
    if (isInput) inputProj(0);
    __syncthreads();

    for (int t = 0; t < S_LEN; ++t) {
        if (isInput) {
            if (t + 1 < S_LEN) inputProj(t + 1);
        } else {
            // ---- wait until every WG finished step t-1 (h_{t-1} visible) ----
            if (t > 0 && lane == 0) {
                const unsigned target = (unsigned)(NWG * t);
                while (__hip_atomic_load(ctr, __ATOMIC_ACQUIRE, __HIP_MEMORY_SCOPE_AGENT) < target)
                    __builtin_amdgcn_s_sleep(1);
            }
            // step t reads parity (t+1)&1, writes parity t&1
            const bf16_t* hrd = hbf + (size_t)((t + 1) & 1) * BATCH * HID + (size_t)Mrow * HID;
            bf16_t*       hwr = hbf + (size_t)(t & 1) * BATCH * HID;

            v8f ar = splat8(bias0), az = splat8(bias1), an = splat8(bias2);
#pragma unroll
            for (int kt = 0; kt < 8; ++kt) {
                const int ka = kt * 32 + hi * 8;
                v8bf c0 = *(const v8bf*)(hrd + ka);
                v8bf c1 = *(const v8bf*)(hrd + ka + 16);
                v16bf a;
#pragma unroll
                for (int i = 0; i < 8; ++i) { a[i] = c0[i]; a[i + 8] = c1[i]; }
                ar = __builtin_amdgcn_wmma_f32_16x16x32_bf16(false, a, false, wt[0 * 8 + kt], (short)0, ar, false, false);
                az = __builtin_amdgcn_wmma_f32_16x16x32_bf16(false, a, false, wt[1 * 8 + kt], (short)0, az, false, false);
                an = __builtin_amdgcn_wmma_f32_16x16x32_bf16(false, a, false, wt[2 * 8 + kt], (short)0, an, false, false);
            }

            // ---- gates + state update (lane-wise on the D layout) ----
            const v8f pr = buf[t & 1][j][0][lane];
            const v8f pz = buf[t & 1][j][1][lane];
            const v8f pn = buf[t & 1][j][2][lane];
            float* outp = out + (size_t)t * BATCH * HID;
#pragma unroll
            for (int e = 0; e < 8; ++e) {
                const float r  = fast_sigmoid(pr[e] + ar[e]);
                const float z  = fast_sigmoid(pz[e] + az[e]);
                const float nn = fast_tanh(pn[e] + r * an[e]);
                const int Me = m0 + e + hi * 8;
                const size_t idx = (size_t)Me * HID + Ncol;
                const float hprev = hf[idx];
                const float hnew  = (1.0f - z) * nn + z * hprev;
                hf[idx]   = hnew;
                hwr[idx]  = (bf16_t)hnew;
                outp[idx] = hnew;
                if (t == S_LEN - 1)
                    out[(size_t)S_LEN * BATCH * HID + idx] = hnew;  // h_n
            }
            __threadfence();  // make h stores agent-visible before the signal
        }

        __syncthreads();  // joins: LDS buf[(t+1)&1] ready; recurrent stores fenced
        if (tid == 0)
            __hip_atomic_fetch_add(ctr, 1u, __ATOMIC_RELEASE, __HIP_MEMORY_SCOPE_AGENT);
    }
}

// ---------------------------------------------------------------------------
// Host-side launch. Workspace layout (bytes):
//   [0,256)            barrier counter
//   [256, +768K)       bf16 weights (Whr,Whz,Whn,Wir,Wiz,Win)
//   [.., +64K)         bf16 h ping-pong (2 x 64x256)
//   [.., +64K)         f32 h
// Total: 917,760 bytes.
// ---------------------------------------------------------------------------
extern "C" void kernel_launch(void* const* d_in, const int* in_sizes, int n_in,
                              void* d_out, int out_size, void* d_ws, size_t ws_size,
                              hipStream_t stream) {
    (void)in_sizes; (void)n_in; (void)out_size; (void)ws_size;

    const float* x    = (const float*)d_in[0];
    const float* Wir  = (const float*)d_in[1];
    const float* Wiz  = (const float*)d_in[2];
    const float* Win  = (const float*)d_in[3];
    const float* Whr  = (const float*)d_in[4];
    const float* Whz  = (const float*)d_in[5];
    const float* Whn  = (const float*)d_in[6];
    const float* b_ir = (const float*)d_in[7];
    const float* b_hr = (const float*)d_in[8];
    const float* b_iz = (const float*)d_in[9];
    const float* b_hz = (const float*)d_in[10];
    const float* b_in = (const float*)d_in[11];
    const float* b_hn = (const float*)d_in[12];

    char* ws = (char*)d_ws;
    unsigned* ctr = (unsigned*)ws;
    bf16_t*   wbf = (bf16_t*)(ws + 256);
    bf16_t*   hbf = (bf16_t*)(ws + 256 + 786432);
    float*    hf  = (float*)(ws + 256 + 786432 + 65536);

    gru_prep<<<dim3(1728), dim3(256), 0, stream>>>(Whr, Whz, Whn, Wir, Wiz, Win,
                                                   wbf, hbf, hf, ctr);
    gru_main<<<dim3(NWG), dim3(256), 0, stream>>>(x, wbf, hbf, hf, ctr,
                                                  b_ir, b_iz, b_in, b_hr, b_hz, b_hn,
                                                  (float*)d_out);
}
